// GCNEncoder_70815420776803
// MI455X (gfx1250) — matrix operand order repaired and checked by reading, not compiled, not run on hardware
//
#include <hip/hip_runtime.h>
#include <hip/hip_bf16.h>

typedef __attribute__((ext_vector_type(16))) __bf16 v16bf;
typedef __attribute__((ext_vector_type(8)))  float  v8f;

#define EPSBN 1e-5f

// ---------------- utility kernels ----------------

__global__ __launch_bounds__(256) void k_fill_f32(float* __restrict__ p, float v, long n) {
    long i = (long)blockIdx.x * blockDim.x + threadIdx.x;
    if (i < n) p[i] = v;
}

__global__ __launch_bounds__(256) void k_deg_count(const long long* __restrict__ dst,
                                                   float* __restrict__ deg, long E) {
    long i = (long)blockIdx.x * blockDim.x + threadIdx.x;
    if (i < E) atomicAdd(&deg[(int)dst[i]], 1.0f);
}

__global__ __launch_bounds__(256) void k_dinv(float* __restrict__ deg, long n) {
    long i = (long)blockIdx.x * blockDim.x + threadIdx.x;
    if (i < n) deg[i] = rsqrtf(deg[i]);   // deg >= 1 always (self-loop)
}

// ---------------- W pre-pack: f32 [K,64] -> bf16 in B-fragment order ----------------
// Packed layout: for ktile t (=k0/32), ntile nb (0..3), lane l (0..31), i (0..15):
//   Bp[(((t*4 + nb)*32 + l)*16) + i] = bf16( W[(t*32 + (i<8 ? (l>>4)*8 + i : 16 + (l>>4)*8 + i - 8))*64 + nb*16 + (l&15)] )
// so each GEMM lane loads its 16 bf16 B values as one contiguous 32B vector.
__global__ __launch_bounds__(256) void k_pack_w(const float* __restrict__ W,
                                                __bf16* __restrict__ Bp, int K) {
    int tid = blockIdx.x * blockDim.x + threadIdx.x;
    int total = K * 64;
    if (tid >= total) return;
    int i  = tid & 15;
    int l  = (tid >> 4) & 31;
    int nb = (tid >> 9) & 3;
    int t  = tid >> 11;
    int hi  = l >> 4;
    int l16 = l & 15;
    int krow = t * 32 + (i < 8 ? hi * 8 + i : 16 + hi * 8 + (i - 8));
    Bp[tid] = (__bf16)W[(long)krow * 64 + nb * 16 + l16];
}

// ---------------- WMMA GEMM: Y[r, 0..63] = dinv[r] * (X[r,:] @ W) ----------------
// One wave computes a full 16-row x 64-col stripe: A fragment loaded once per
// k-step, reused by 4 back-to-back WMMAs. K templated so loop fully unrolls.
template <int K>
__global__ __launch_bounds__(256) void k_gemm_wmma(const float* __restrict__ X,
                                                   const __bf16* __restrict__ Bp,
                                                   const float* __restrict__ dinv,
                                                   float* __restrict__ Y, int N) {
    int wave = threadIdx.x >> 5;
    int lane = threadIdx.x & 31;
    int hi   = lane >> 4;
    int l16  = lane & 15;

    int tiles_m = (N + 15) >> 4;
    int mt = blockIdx.x * 8 + wave;
    if (mt >= tiles_m) return;

    int rowA = mt * 16 + l16;
    if (rowA > N - 1) rowA = N - 1;    // clamp reads for the ragged tail tile

    v8f acc0 = {}, acc1 = {}, acc2 = {}, acc3 = {};

    #pragma unroll
    for (int t = 0; t < K / 32; ++t) {
        const int k0 = t * 32;
        // ---- A fragment (16-bit 16x32 layout): two 8-wide K runs per lane ----
        const float* xr = X + (long)rowA * K + k0 + hi * 8;
        float4 p0 = *(const float4*)(xr + 0);
        float4 p1 = *(const float4*)(xr + 4);
        float4 p2 = *(const float4*)(xr + 16);
        float4 p3 = *(const float4*)(xr + 20);
        v16bf a;
        a[0]  = (__bf16)p0.x; a[1]  = (__bf16)p0.y; a[2]  = (__bf16)p0.z; a[3]  = (__bf16)p0.w;
        a[4]  = (__bf16)p1.x; a[5]  = (__bf16)p1.y; a[6]  = (__bf16)p1.z; a[7]  = (__bf16)p1.w;
        a[8]  = (__bf16)p2.x; a[9]  = (__bf16)p2.y; a[10] = (__bf16)p2.z; a[11] = (__bf16)p2.w;
        a[12] = (__bf16)p3.x; a[13] = (__bf16)p3.y; a[14] = (__bf16)p3.z; a[15] = (__bf16)p3.w;

        // ---- B fragments: contiguous 32B per lane per n-tile (pre-packed) ----
        const __bf16* bbase = Bp + ((long)(t * 4) * 32 + lane) * 16;
        v16bf b0 = *(const v16bf*)(bbase + 0 * 32 * 16);
        v16bf b1 = *(const v16bf*)(bbase + 1 * 32 * 16);
        v16bf b2 = *(const v16bf*)(bbase + 2 * 32 * 16);
        v16bf b3 = *(const v16bf*)(bbase + 3 * 32 * 16);

        acc0 = __builtin_amdgcn_wmma_f32_16x16x32_bf16(false, a, false, b0, (short)0, acc0, false, false);
        acc1 = __builtin_amdgcn_wmma_f32_16x16x32_bf16(false, a, false, b1, (short)0, acc1, false, false);
        acc2 = __builtin_amdgcn_wmma_f32_16x16x32_bf16(false, a, false, b2, (short)0, acc2, false, false);
        acc3 = __builtin_amdgcn_wmma_f32_16x16x32_bf16(false, a, false, b3, (short)0, acc3, false, false);
    }

    // ---- epilogue: row r = mt*16 + hi*8 + j, col = nb*16 + l16 ----
    int row0 = mt * 16 + hi * 8;
    if (mt * 16 + 16 <= N) {                 // wave-uniform fast path, branch-free stores
        float dv[8];
        #pragma unroll
        for (int j = 0; j < 8; ++j) dv[j] = dinv[row0 + j];
        float* yr = Y + (long)row0 * 64 + l16;
        #pragma unroll
        for (int j = 0; j < 8; ++j) {
            yr[(long)j * 64 + 0]  = dv[j] * acc0[j];
            yr[(long)j * 64 + 16] = dv[j] * acc1[j];
            yr[(long)j * 64 + 32] = dv[j] * acc2[j];
            yr[(long)j * 64 + 48] = dv[j] * acc3[j];
        }
    } else {                                  // ragged tail tile
        #pragma unroll
        for (int j = 0; j < 8; ++j) {
            int r = row0 + j;
            if (r < N) {
                float dv = dinv[r];
                float* yr = Y + (long)r * 64 + l16;
                yr[0]  = dv * acc0[j];
                yr[16] = dv * acc1[j];
                yr[32] = dv * acc2[j];
                yr[48] = dv * acc3[j];
            }
        }
    }
}

// ---------------- edge scatter: agg[dst] += y[src], 16 threads/edge, float4 ----------------
__global__ __launch_bounds__(256) void k_scatter(const long long* __restrict__ src,
                                                 const long long* __restrict__ dst,
                                                 const float* __restrict__ y,
                                                 float* __restrict__ agg, long E) {
    long t = (long)blockIdx.x * blockDim.x + threadIdx.x;
    long e = t >> 4;
    if (e >= E) return;
    int c4 = ((int)t & 15) * 4;
    int s = (int)src[e], d = (int)dst[e];
    const float4 v = *(const float4*)(y + (long)s * 64 + c4);
    float* o = agg + (long)d * 64 + c4;
    atomicAdd(o + 0, v.x);
    atomicAdd(o + 1, v.y);
    atomicAdd(o + 2, v.z);
    atomicAdd(o + 3, v.w);
}

// ---------------- h = relu(BN(dinv*(agg + y) + bias)) ----------------
__global__ __launch_bounds__(256) void k_finalize(const float* __restrict__ agg,
                                                  const float* __restrict__ y,
                                                  const float* __restrict__ dinv,
                                                  const float* __restrict__ bias,
                                                  const float* __restrict__ gamma,
                                                  const float* __restrict__ beta,
                                                  const float* __restrict__ rm,
                                                  const float* __restrict__ rv,
                                                  float* __restrict__ h, long n) {
    long i = (long)blockIdx.x * blockDim.x + threadIdx.x;
    if (i >= n) return;
    int c = (int)(i & 63);
    long nd = i >> 6;
    float v = dinv[nd] * (agg[i] + y[i]) + bias[c];
    v = (v - rm[c]) * rsqrtf(rv[c] + EPSBN) * gamma[c] + beta[c];
    h[i] = fmaxf(v, 0.0f);
}

// ---------------- global mean pool ----------------
__global__ __launch_bounds__(256) void k_pool_acc(const float* __restrict__ h,
                                                  const long long* __restrict__ b,
                                                  float* __restrict__ sums,
                                                  float* __restrict__ cnt, long n) {
    long i = (long)blockIdx.x * blockDim.x + threadIdx.x;
    if (i >= n) return;
    int c = (int)(i & 63);
    long nd = i >> 6;
    int g = (int)b[nd];
    atomicAdd(&sums[(long)g * 64 + c], h[i]);
    if (c == 0) atomicAdd(&cnt[g], 1.0f);
}

__global__ __launch_bounds__(256) void k_pool_div(const float* __restrict__ sums,
                                                  const float* __restrict__ cnt,
                                                  float* __restrict__ out, long n) {
    long i = (long)blockIdx.x * blockDim.x + threadIdx.x;
    if (i >= n) return;
    out[i] = sums[i] / fmaxf(cnt[i >> 6], 1.0f);
}

// ---------------- host launch ----------------

static inline unsigned gcnv_blocks(long n, int bs) { return (unsigned)((n + bs - 1) / bs); }

extern "C" void kernel_launch(void* const* d_in, const int* in_sizes, int n_in,
                              void* d_out, int out_size, void* d_ws, size_t ws_size,
                              hipStream_t stream) {
    const float*     x   = (const float*)d_in[0];
    const long long* e   = (const long long*)d_in[1];    // [2, E] int64
    const long long* b   = (const long long*)d_in[2];    // [N] int64
    const float*     W1  = (const float*)d_in[4];
    const float*     b1  = (const float*)d_in[5];
    const float*     g1  = (const float*)d_in[6];
    const float*     be1 = (const float*)d_in[7];
    const float*     rm1 = (const float*)d_in[8];
    const float*     rv1 = (const float*)d_in[9];
    const float*     W2  = (const float*)d_in[10];
    const float*     b2  = (const float*)d_in[11];
    const float*     g2  = (const float*)d_in[12];
    const float*     be2 = (const float*)d_in[13];
    const float*     rm2 = (const float*)d_in[14];
    const float*     rv2 = (const float*)d_in[15];
    float*           out = (float*)d_out;

    const int  D = 128, H = 64;
    const long N = (long)in_sizes[0] / D;
    const long E = (long)in_sizes[1] / 2;
    const long G = (long)out_size / H;
    const long long* esrc = e;
    const long long* edst = e + E;

    // workspace layout (floats), bf16 pack buffers 32B-aligned at the end
    float* ws   = (float*)d_ws;
    float* dinv = ws;                 // [N]
    float* y    = dinv + N;           // [N,64]
    float* agg  = y + N * 64;         // [N,64]
    float* h    = agg + N * 64;       // [N,64]
    float* sums = h + N * 64;         // [G,64]
    float* cnt  = sums + G * 64;      // [G]
    uintptr_t pk = (uintptr_t)(cnt + G);
    pk = (pk + 31) & ~(uintptr_t)31;
    __bf16* bp1 = (__bf16*)pk;        // [128*64] packed bf16 W1
    __bf16* bp2 = bp1 + 128 * 64;     // [64*64]  packed bf16 W2

    const int BS = 256;
    const long NH = N * 64;
    const unsigned gemm_blocks = (unsigned)(((N + 15) / 16 + 7) / 8);   // 8 waves/block

    // --- degree / dinv (shared by both layers) ---
    k_fill_f32<<<gcnv_blocks(N, BS), BS, 0, stream>>>(dinv, 1.0f, N);  // self-loop
    k_deg_count<<<gcnv_blocks(E, BS), BS, 0, stream>>>(edst, dinv, E);
    k_dinv<<<gcnv_blocks(N, BS), BS, 0, stream>>>(dinv, N);

    // --- pack both weight matrices (tiny) ---
    k_pack_w<<<gcnv_blocks(128 * 64, BS), BS, 0, stream>>>(W1, bp1, 128);
    k_pack_w<<<gcnv_blocks(64 * 64, BS), BS, 0, stream>>>(W2, bp2, 64);

    // --- layer 1 ---
    k_gemm_wmma<128><<<gemm_blocks, BS, 0, stream>>>(x, bp1, dinv, y, (int)N);
    k_fill_f32<<<gcnv_blocks(NH, BS), BS, 0, stream>>>(agg, 0.0f, NH);
    k_scatter<<<gcnv_blocks(E * 16, BS), BS, 0, stream>>>(esrc, edst, y, agg, E);
    k_finalize<<<gcnv_blocks(NH, BS), BS, 0, stream>>>(agg, y, dinv, b1, g1, be1, rm1, rv1, h, NH);

    // --- layer 2 (reuse y/agg buffers) ---
    k_gemm_wmma<64><<<gemm_blocks, BS, 0, stream>>>(h, bp2, dinv, y, (int)N);
    k_fill_f32<<<gcnv_blocks(NH, BS), BS, 0, stream>>>(agg, 0.0f, NH);
    k_scatter<<<gcnv_blocks(E * 16, BS), BS, 0, stream>>>(esrc, edst, y, agg, E);
    k_finalize<<<gcnv_blocks(NH, BS), BS, 0, stream>>>(agg, y, dinv, b2, g2, be2, rm2, rv2, h, NH);

    // --- global mean pool ---
    k_fill_f32<<<gcnv_blocks(G * 64, BS), BS, 0, stream>>>(sums, 0.0f, G * 64);
    k_fill_f32<<<gcnv_blocks(G, BS), BS, 0, stream>>>(cnt, 0.0f, G);
    k_pool_acc<<<gcnv_blocks(NH, BS), BS, 0, stream>>>(h, b, sums, cnt, NH);
    k_pool_div<<<gcnv_blocks(G * 64, BS), BS, 0, stream>>>(sums, cnt, out, G * 64);
}
